// LSTM_3590592659648
// MI455X (gfx1250) — compile-verified, matrix operand order checked by reading, not run on hardware
//
#include <hip/hip_runtime.h>

typedef __bf16 bf16x16 __attribute__((ext_vector_type(16)));
typedef float  f32x8   __attribute__((ext_vector_type(8)));

#define T_STEPS 512
#define BATCH   64
#define DIN     256
#define DH      512
#define NWG     32
#define TPB     128

__device__ __forceinline__ unsigned short f2bf(float f) {
  unsigned int u = __float_as_uint(f);
  unsigned int r = u + 0x7FFFu + ((u >> 16) & 1u);   // round-to-nearest-even
  return (unsigned short)(r >> 16);
}

// Load one 16-element bf16 fragment: elements [0..7] at p, [8..15] at p+16
// (matches the CDNA5 16-bit A/B per-lane K pattern: K = 16*(j>>3) + half*8 + (j&7))
__device__ __forceinline__ bf16x16 ldfrag(const unsigned short* p) {
  union { uint4 q[2]; bf16x16 v; } r;
  r.q[0] = *(const uint4*)(p);
  r.q[1] = *(const uint4*)(p + 16);
  return r.v;
}

__device__ __forceinline__ void gbar(int* cnt, int* gen) {
  __syncthreads();
  if (threadIdx.x == 0) {
    __threadfence();
    int g = __hip_atomic_load(gen, __ATOMIC_RELAXED, __HIP_MEMORY_SCOPE_AGENT);
    int p = __hip_atomic_fetch_add(cnt, 1, __ATOMIC_ACQ_REL, __HIP_MEMORY_SCOPE_AGENT);
    if (p == NWG - 1) {
      __hip_atomic_store(cnt, 0, __ATOMIC_RELAXED, __HIP_MEMORY_SCOPE_AGENT);
      __hip_atomic_fetch_add(gen, 1, __ATOMIC_RELEASE, __HIP_MEMORY_SCOPE_AGENT);
    } else {
      while (__hip_atomic_load(gen, __ATOMIC_ACQUIRE, __HIP_MEMORY_SCOPE_AGENT) == g) {
        __builtin_amdgcn_s_sleep(4);
      }
    }
  }
  __syncthreads();
}

// ---------------- prep: fp32 -> bf16, build transposed (B^T) weights ----------------
__global__ void lstm_prep(const float* __restrict__ X,
                          const float* __restrict__ Wfx, const float* __restrict__ Wfg, const float* __restrict__ bfg,
                          const float* __restrict__ Wix, const float* __restrict__ Wig, const float* __restrict__ big,
                          const float* __restrict__ Wux, const float* __restrict__ Wug, const float* __restrict__ bug,
                          const float* __restrict__ Wox, const float* __restrict__ Wog, const float* __restrict__ bog,
                          unsigned short* __restrict__ xb,
                          unsigned short* __restrict__ wxt,
                          unsigned short* __restrict__ wht,
                          int* cnt, int* gen) {
  const long NX  = (long)T_STEPS * BATCH * DIN;   // 8,388,608
  const long NWX = 4L * DH * DIN;                 //   524,288
  const long NWH = 4L * DH * DH;                  // 1,048,576
  long gid    = (long)blockIdx.x * blockDim.x + threadIdx.x;
  long stride = (long)gridDim.x * blockDim.x;
  for (long i = gid; i < NX + NWX + NWH; i += stride) {
    if (i < NX) {
      xb[i] = f2bf(X[i]);
    } else if (i < NX + NWX) {
      long j = i - NX;
      int p = (int)(j >> 8), k = (int)(j & (DIN - 1));
      int g = p >> 9, nn = p & (DH - 1);
      const float* W = (g == 0) ? Wfx : (g == 1) ? Wix : (g == 2) ? Wux : Wox;
      wxt[j] = f2bf(W[(long)k * DH + nn]);        // wxt[(g*512+n)*256 + k] = Wx[k][n]
    } else {
      long j = i - NX - NWX;
      int p = (int)(j >> 9), k = (int)(j & (DH - 1));
      int g = p >> 9, nn = p & (DH - 1);
      const float* W  = (g == 0) ? Wfg : (g == 1) ? Wig : (g == 2) ? Wug : Wog;
      const float* bb = (g == 0) ? bfg : (g == 1) ? big : (g == 2) ? bug : bog;
      // reference adds bias to the weight MATRIX: (W_g + b_g)[k][n] = W[k][n] + b[n]
      wht[j] = f2bf(W[(long)k * DH + nn] + bb[nn]);
    }
  }
  if (gid == 0) { *cnt = 0; *gen = 0; }           // barrier state (ws may be poisoned)
}

// ---------------- persistent fused LSTM recurrence ----------------
// Grid: 32 WGs x 128 threads. WG = one 16-wide H column tile (weights in LDS),
// waves 0..3 = batch tiles of 16. Each wave owns a 16x16 (batch x H) tile and
// keeps c in registers across all 512 steps. One atomic grid barrier per step;
// h double-buffered in global (bf16, L2 resident).
__global__ void __launch_bounds__(TPB, 1) lstm_recurrent(
    const unsigned short* __restrict__ xb,
    const unsigned short* __restrict__ wxt,
    const unsigned short* __restrict__ wht,
    unsigned short* __restrict__ hb0,
    unsigned short* __restrict__ hb1,
    float* __restrict__ out,
    int* cnt, int* gen) {
  __shared__ unsigned short lds[4 * 16 * DIN + 4 * 16 * DH];  // 96 KB (of 320 KB/WGP)

  const int tid   = threadIdx.x;
  const int lane  = tid & 31;
  const int mtile = tid >> 5;        // 0..3
  const int ht    = blockIdx.x;      // 0..31  -> H columns [ht*16, ht*16+16)
  const int n     = lane & 15;       // A row / B col / D col within tile
  const int hf    = lane >> 4;       // lane half selects K sub-pattern

  // zero initial h buffer (h0 = 0)
  for (int i = blockIdx.x * TPB + tid; i < BATCH * DH; i += NWG * TPB) hb0[i] = 0;

  // stage this WG's weight slice (all 4 gates) into LDS, B^T layout, K contiguous
  for (int i = tid; i < 4 * 16 * DIN; i += TPB) {
    int pr = i >> 8, k = i & (DIN - 1);
    int g = pr >> 4, c = pr & 15;
    lds[i] = wxt[((long)(g * DH + ht * 16 + c)) * DIN + k];
  }
  for (int i = tid; i < 4 * 16 * DH; i += TPB) {
    int pr = i >> 9, k = i & (DH - 1);
    int g = pr >> 4, c = pr & 15;
    lds[4 * 16 * DIN + i] = wht[((long)(g * DH + ht * 16 + c)) * DH + k];
  }
  __syncthreads();
  gbar(cnt, gen);

  float creg[8];
  #pragma unroll
  for (int v = 0; v < 8; ++v) creg[v] = 0.0f;

  const f32x8 vzero = {0.f, 0.f, 0.f, 0.f, 0.f, 0.f, 0.f, 0.f};
  const int arow = mtile * 16 + n;   // A-fragment batch row for this lane

  for (int t = 0; t < T_STEPS; ++t) {
    const unsigned short* hread  = (t & 1) ? hb1 : hb0;
    unsigned short*       hwrite = (t & 1) ? hb0 : hb1;

    f32x8 acc[4];
    #pragma unroll
    for (int g = 0; g < 4; ++g) acc[g] = vzero;

    // ---- input-side K (256): g += X_t @ Wx (fused, no Xg intermediate) ----
    const unsigned short* ax = xb + ((long)t * BATCH + arow) * DIN;
    #pragma unroll
    for (int kc = 0; kc < DIN / 32; ++kc) {
      const int ko = kc * 32 + hf * 8;
      bf16x16 a = ldfrag(ax + ko);
      #pragma unroll
      for (int g = 0; g < 4; ++g) {
        bf16x16 b = ldfrag(&lds[(g * 16 + n) * DIN + ko]);
        acc[g] = __builtin_amdgcn_wmma_f32_16x16x32_bf16(
            false, a, false, b, (short)0, acc[g], false, false);
      }
    }
    // ---- hidden-side K (512): g += h_{t-1} @ (Wh + b) ----
    const unsigned short* ah = hread + arow * DH;
    #pragma unroll
    for (int kc = 0; kc < DH / 32; ++kc) {
      const int ko = kc * 32 + hf * 8;
      bf16x16 a = ldfrag(ah + ko);
      #pragma unroll
      for (int g = 0; g < 4; ++g) {
        bf16x16 b = ldfrag(&lds[4 * 16 * DIN + (g * 16 + n) * DH + ko]);
        acc[g] = __builtin_amdgcn_wmma_f32_16x16x32_bf16(
            false, a, false, b, (short)0, acc[g], false, false);
      }
    }

    // ---- gates + state update; D layout: col = n, row = mtile*16 + hf*8 + v ----
    const int dcol  = ht * 16 + n;
    const int mbase = mtile * 16 + hf * 8;
    #pragma unroll
    for (int v = 0; v < 8; ++v) {
      float fg = 1.0f / (1.0f + __expf(-acc[0][v]));
      float ig = 1.0f / (1.0f + __expf(-acc[1][v]));
      float ug = tanhf(acc[2][v]);
      float og = 1.0f / (1.0f + __expf(-acc[3][v]));
      float cn = creg[v] * fg + ig * ug;
      creg[v] = cn;
      float hn = og * tanhf(cn);
      const int m = mbase + v;
      out[((long)t * BATCH + m) * DH + dcol] = hn;            // outs (T*B, H)
      hwrite[m * DH + dcol] = f2bf(hn);
      if (t == T_STEPS - 1) {
        out[(long)T_STEPS * BATCH * DH + m * DH + dcol] = hn;                 // final h
        out[(long)T_STEPS * BATCH * DH + BATCH * DH + m * DH + dcol] = cn;    // final c
      }
    }
    gbar(cnt, gen);   // h_t visible to all WGs before step t+1
  }
}

extern "C" void kernel_launch(void* const* d_in, const int* in_sizes, int n_in,
                              void* d_out, int out_size, void* d_ws, size_t ws_size,
                              hipStream_t stream) {
  const float* X   = (const float*)d_in[0];
  const float* Wfx = (const float*)d_in[1];
  const float* Wfg = (const float*)d_in[2];
  const float* bfg = (const float*)d_in[3];
  const float* Wix = (const float*)d_in[4];
  const float* Wig = (const float*)d_in[5];
  const float* big = (const float*)d_in[6];
  const float* Wux = (const float*)d_in[7];
  const float* Wug = (const float*)d_in[8];
  const float* bug = (const float*)d_in[9];
  const float* Wox = (const float*)d_in[10];
  const float* Wog = (const float*)d_in[11];
  const float* bog = (const float*)d_in[12];
  float* out = (float*)d_out;

  // workspace layout (all 16B aligned)
  char* ws = (char*)d_ws;
  int* cnt = (int*)(ws + 0);
  int* gen = (int*)(ws + 4);
  unsigned short* hb0 = (unsigned short*)(ws + 256);                      // 64 KB
  unsigned short* hb1 = (unsigned short*)(ws + 256 + 65536);              // 64 KB
  unsigned short* wxt = (unsigned short*)(ws + 256 + 2 * 65536);          // 1 MB
  unsigned short* wht = (unsigned short*)(ws + 256 + 2 * 65536 + 1048576);// 2 MB
  unsigned short* xb  = (unsigned short*)(ws + 256 + 2 * 65536 + 1048576 + 2097152); // 16 MB

  lstm_prep<<<4096, 256, 0, stream>>>(X, Wfx, Wfg, bfg, Wix, Wig, big,
                                      Wux, Wug, bug, Wox, Wog, bog,
                                      xb, wxt, wht, cnt, gen);
  lstm_recurrent<<<NWG, TPB, 0, stream>>>(xb, wxt, wht, hb0, hb1, out, cnt, gen);
}
